// Attention_70274254897356
// MI455X (gfx1250) — matrix operand order repaired and checked by reading, not compiled, not run on hardware
//
#include <hip/hip_runtime.h>
#include <math.h>

typedef __bf16 bf16;
typedef __attribute__((ext_vector_type(16))) __bf16 v16bf;
typedef __attribute__((ext_vector_type(8)))  __bf16 v8bf;
typedef __attribute__((ext_vector_type(4)))  __bf16 v4bf;
typedef __attribute__((ext_vector_type(8)))  float  v8f;
typedef __attribute__((ext_vector_type(4)))  float  v4f;
typedef __attribute__((ext_vector_type(4)))  int    v4i;
typedef __attribute__((ext_vector_type(8)))  int    v8i;
typedef __attribute__((ext_vector_type(4)))  unsigned int v4u;

// Address-space-qualified types (AS1 = global, AS3 = LDS)
typedef __attribute__((address_space(1))) v4i gas_v4i;
typedef __attribute__((address_space(3))) v4i las_v4i;
typedef __attribute__((address_space(3))) char as3char;

#define BDIM 4
#define SDIM 4096
#define DMODEL 2048
#define DATTN 2048

// ---- CDNA5 data-mover feature detection --------------------------------
#if defined(__has_builtin)
#if __has_builtin(__builtin_amdgcn_global_load_async_to_lds_b128) && \
    __has_builtin(__builtin_amdgcn_s_wait_asynccnt)
#define USE_ASYNC 1
#endif
#if __has_builtin(__builtin_amdgcn_tensor_load_to_lds) && \
    __has_builtin(__builtin_amdgcn_s_wait_tensorcnt)
#define USE_TDM 1
#if __has_include(<hip/amd_detail/amd_gfx1250_TDM.h>)
#define TDM_6ARG 1      // therock-10.0 headers -> 6-arg builtin
#else
#define TDM_6ARG 0      // ROCm 7.2 -> 5-arg builtin
#endif
#endif
#endif
#ifndef USE_ASYNC
#define USE_ASYNC 0
#endif
#ifndef USE_TDM
#define USE_TDM 0
#endif

constexpr int BM = 128, BN = 256, BK = 32;
constexpr int LDSS = BK + 8;      // padded LDS row stride in halves (80B rows)

// LDS byte offset of a generic pointer into shared memory
__device__ __forceinline__ unsigned lds_off(const void* p) {
  return (unsigned)(unsigned long long)(as3char*)(char*)(void*)p;
}

#if USE_TDM
// ---- Tensor DMA descriptor construction (ISA 08_async_tensor §8) --------
// group0: [1:0]=count=1, [63:32]=lds_addr, [120:64]=global_addr, [127:126]=type=2
__device__ __forceinline__ v4u tdm_g0(unsigned lds_addr, unsigned long long gaddr) {
  v4u g;
  g[0] = 1u;
  g[1] = lds_addr;
  g[2] = (unsigned)gaddr;
  g[3] = (unsigned)((gaddr >> 32) & 0x01FFFFFFull) | (2u << 30);
  return g;
}
// group1: data_size=1 (2B), pad_enable=1, pad_interval=3 (16 DW = 64B rows),
// pad_amount=3 (4 DW = 16B) -> 80B padded LDS rows == LDSS*2.
// tensor_dim0 = rowlen, tensor_dim1 = rows, tile_dim0 = BK, tile_dim1 = rows,
// tensor_dim0_stride = rowlen.
__device__ __forceinline__ v8i tdm_g1(int rows, int rowlen) {
  const unsigned w0 = (1u << 16) | (1u << 20) | (3u << 22) | (3u << 25);
  const unsigned td0 = (unsigned)rowlen, td1 = (unsigned)rows;
  v8i g;
  g[0] = (int)w0;
  g[1] = (int)((td0 & 0xFFFFu) << 16);                       // td0[15:0] @ [63:48]
  g[2] = (int)(((td0 >> 16) & 0xFFFFu) | ((td1 & 0xFFFFu) << 16));
  g[3] = (int)(((td1 >> 16) & 0xFFFFu) | ((unsigned)BK << 16)); // tile_dim0
  g[4] = (int)(td1 & 0xFFFFu);                                // tile_dim1 (tile_dim2=0)
  g[5] = (int)td0;                                            // dim0_stride[31:0]
  g[6] = 0;                                                   // dim0_stride[47:32]
  g[7] = 0;
  return g;
}
__device__ __forceinline__ void tdm_load(unsigned ldst, const bf16* g, int rows, int rowlen) {
  v4i z4 = {};
#if TDM_6ARG
  v8i z8 = {};
  __builtin_amdgcn_tensor_load_to_lds(tdm_g0(ldst, (unsigned long long)(uintptr_t)g),
                                      tdm_g1(rows, rowlen), z4, z4, z8, 0);
#else
  __builtin_amdgcn_tensor_load_to_lds(tdm_g0(ldst, (unsigned long long)(uintptr_t)g),
                                      tdm_g1(rows, rowlen), z4, z4, 0);
#endif
}
#endif

__device__ __forceinline__ void cp16(const bf16* g, bf16* l) {
#if USE_ASYNC
  __builtin_amdgcn_global_load_async_to_lds_b128(
      (gas_v4i*)(v4i*)(void*)g, (las_v4i*)(v4i*)(void*)l, 0, 0);
#else
  *(v8bf*)l = *(const v8bf*)g;
#endif
}

// ---------------------------------------------------------------- convert ---
__global__ __launch_bounds__(256) void cvt_f32_bf16(const float* __restrict__ in,
                                                    bf16* __restrict__ out, long n4) {
  long i = (long)blockIdx.x * blockDim.x + threadIdx.x;
  if (i >= n4) return;
  v4f v = *(const v4f*)(in + i * 4);
  v4bf o;
  #pragma unroll
  for (int j = 0; j < 4; ++j) o[j] = (bf16)v[j];
  *(v4bf*)(out + i * 4) = o;
}

// ------------------------------------------------------------------- GEMM ---
// C = scale * (A @ B^T) (+bias).  A:[M,K] lda, B:[N,K] ldb (K contiguous both).
// OUT_T: store C transposed as bf16 (C^T[col*ldc + row], 16B vector stores).
template <bool OUT_F32, bool HAS_BIAS, bool OUT_T>
__global__ __launch_bounds__(256) void gemm_wmma(
    const bf16* __restrict__ A, int lda, const bf16* __restrict__ Bm, int ldb,
    void* __restrict__ Cv, int ldc, const float* __restrict__ bias,
    int N, int K, float scale, long strideA, long strideB, long strideC) {
  __shared__ bf16 As[2][BM][LDSS];
  __shared__ bf16 Bs[2][BN][LDSS];

  const int tid  = threadIdx.x;
  const int lane = tid & 31;
  const int wid  = tid >> 5;
  const int wm   = wid & 1;       // 2 M-subtiles of 64 rows
  const int wn   = wid >> 1;      // 4 N-subtiles of 64 cols
  const int rl   = lane & 15;
  const bool hi  = lane >= 16;

  const long zA = (long)blockIdx.z * strideA;
  const long zB = (long)blockIdx.z * strideB;
  const long zC = (long)blockIdx.z * strideC;
  const int m_blk = blockIdx.y * BM;
  const int n_blk = blockIdx.x * BN;

  const bf16* Ab = A + zA + (long)m_blk * lda;
  const bf16* Bb = Bm + zB + (long)n_blk * ldb;

  // ---- tile staging: TDM (1 instr/tile, wave0) > async DMA > sync copy ----
  auto stage = [&](int sb, int k0) {
#if USE_TDM
    if (wid == 0) {
      tdm_load(lds_off(&As[sb][0][0]), Ab + k0, BM, lda);
      tdm_load(lds_off(&Bs[sb][0][0]), Bb + k0, BN, ldb);
    }
#else
    #pragma unroll
    for (int it = 0; it < 2; ++it) {              // A tile: 128x32 halves
      int idx = (tid + it * 256) * 8;
      int r = idx >> 5, c = idx & 31;
      cp16(Ab + (long)r * lda + k0 + c, &As[sb][r][c]);
    }
    #pragma unroll
    for (int it = 0; it < 4; ++it) {              // B tile: 256x32 halves
      int idx = (tid + it * 256) * 8;
      int r = idx >> 5, c = idx & 31;
      cp16(Bb + (long)r * ldb + k0 + c, &Bs[sb][r][c]);
    }
#endif
  };
  auto wait_more = [&]() {                         // next stage in flight
#if USE_TDM
    if (wid == 0) __builtin_amdgcn_s_wait_tensorcnt(2);
#elif USE_ASYNC
    __builtin_amdgcn_s_wait_asynccnt(6);
#endif
  };
  auto wait_last = [&]() {                         // nothing else in flight
#if USE_TDM
    if (wid == 0) __builtin_amdgcn_s_wait_tensorcnt(0);
#elif USE_ASYNC
    __builtin_amdgcn_s_wait_asynccnt(0);
#endif
  };

  v8f acc[4][4] = {};

  auto compute = [&](int buf) {
    v16bf af[4], bfr[4];
    const int abq = hi ? 8 : 0;                    // A: lanes16-31 hold K 8..15/24..31
    #pragma unroll
    for (int i = 0; i < 4; ++i) {
      int r = wm * 64 + i * 16 + rl;
      v8bf lo8 = *(const v8bf*)&As[buf][r][abq];
      v8bf hi8 = *(const v8bf*)&As[buf][r][abq + 16];
      #pragma unroll
      for (int j = 0; j < 8; ++j) { af[i][j] = lo8[j]; af[i][j + 8] = hi8[j]; }
    }
    const int bbq = hi ? 8 : 0;                    // B: lanes16-31 hold K 8..15/24..31
    #pragma unroll
    for (int i = 0; i < 4; ++i) {
      int c = wn * 64 + i * 16 + rl;
      v8bf lo8 = *(const v8bf*)&Bs[buf][c][bbq];
      v8bf hi8 = *(const v8bf*)&Bs[buf][c][bbq + 16];
      #pragma unroll
      for (int j = 0; j < 8; ++j) { bfr[i][j] = lo8[j]; bfr[i][j + 8] = hi8[j]; }
    }
    #pragma unroll
    for (int i = 0; i < 4; ++i)
      #pragma unroll
      for (int j = 0; j < 4; ++j)
        acc[i][j] = __builtin_amdgcn_wmma_f32_16x16x32_bf16(
            false, af[i], false, bfr[j], (short)0, acc[i][j], false, false);
  };

  // ---- software-pipelined main loop (peeled tail: no branches in body) ----
  stage(0, 0);
  int buf = 0;
  int k0 = 0;
  for (; k0 + BK < K; k0 += BK) {
    stage(buf ^ 1, k0 + BK);       // DMA next stage while computing current
    wait_more();
    __syncthreads();               // current stage visible to all waves
    compute(buf);
    __syncthreads();               // all reads done; next DMA may land
    buf ^= 1;
  }
  wait_last();
  __syncthreads();
  compute(buf);

  // ---- epilogue: C layout = VGPR r -> (M=r | M=r+8), N=lane&15 ----
  #pragma unroll
  for (int i = 0; i < 4; ++i) {
    int mrow0 = m_blk + wm * 64 + i * 16 + (hi ? 8 : 0);
    #pragma unroll
    for (int j = 0; j < 4; ++j) {
      int col = n_blk + wn * 64 + j * 16 + rl;
      if (OUT_T) {
        v8bf o;
        #pragma unroll
        for (int r = 0; r < 8; ++r) o[r] = (bf16)(acc[i][j][r] * scale);
        *(v8bf*)((bf16*)Cv + zC + (long)col * ldc + mrow0) = o;
      } else {
        #pragma unroll
        for (int r = 0; r < 8; ++r) {
          float val = acc[i][j][r] * scale;
          long off = zC + (long)(mrow0 + r) * ldc + col;
          if (OUT_F32) {
            if (HAS_BIAS) val += bias[col];
            ((float*)Cv)[off] = val;
          } else {
            ((bf16*)Cv)[off] = (bf16)val;
          }
        }
      }
    }
  }
}

// ---------------------------------------------------------------- softmax ---
__global__ __launch_bounds__(256) void softmax_rows(bf16* __restrict__ P) {
  long row = blockIdx.x;
  bf16* p = P + row * (long)SDIM;
  const int tid = threadIdx.x;
  float v[16];
  v8bf a = *(const v8bf*)(p + tid * 16);
  v8bf b = *(const v8bf*)(p + tid * 16 + 8);
  #pragma unroll
  for (int i = 0; i < 8; ++i) { v[i] = (float)a[i]; v[i + 8] = (float)b[i]; }

  float m = -1e30f;
  #pragma unroll
  for (int i = 0; i < 16; ++i) m = fmaxf(m, v[i]);
  __shared__ float red[256];
  red[tid] = m; __syncthreads();
  for (int s = 128; s > 0; s >>= 1) { if (tid < s) red[tid] = fmaxf(red[tid], red[tid + s]); __syncthreads(); }
  m = red[0]; __syncthreads();

  float sum = 0.f;
  #pragma unroll
  for (int i = 0; i < 16; ++i) { v[i] = __expf(v[i] - m); sum += v[i]; }
  red[tid] = sum; __syncthreads();
  for (int s = 128; s > 0; s >>= 1) { if (tid < s) red[tid] += red[tid + s]; __syncthreads(); }
  float inv = 1.f / red[0];

  v8bf oa, ob;
  #pragma unroll
  for (int i = 0; i < 8; ++i) { oa[i] = (bf16)(v[i] * inv); ob[i] = (bf16)(v[i + 8] * inv); }
  *(v8bf*)(p + tid * 16)     = oa;
  *(v8bf*)(p + tid * 16 + 8) = ob;
}

// ----------------------------------------------------------------- launch ---
extern "C" void kernel_launch(void* const* d_in, const int* in_sizes, int n_in,
                              void* d_out, int out_size, void* d_ws, size_t ws_size,
                              hipStream_t stream) {
  const float* x  = (const float*)d_in[0];
  const float* Wq = (const float*)d_in[1];
  const float* Wk = (const float*)d_in[2];
  const float* Wv = (const float*)d_in[3];
  const float* Wo = (const float*)d_in[4];
  const float* bo = (const float*)d_in[5];

  const long T  = (long)BDIM * SDIM;          // 16384 tokens
  const long TD = T * DMODEL;
  const long WW = (long)DATTN * DMODEL;
  const long SS = (long)SDIM * SDIM;          // per-batch logits
  const long SD = (long)SDIM * DATTN;

  char* w = (char*)d_ws;
  const size_t MB = 1u << 20;
  bf16* xh  = (bf16*)(w);                 // 64 MB (reused as attn_out)
  bf16* wqh = (bf16*)(w + 64  * MB);
  bf16* wkh = (bf16*)(w + 72  * MB);
  bf16* wvh = (bf16*)(w + 80  * MB);
  bf16* woh = (bf16*)(w + 88  * MB);
  bf16* qh  = (bf16*)(w + 96  * MB);      // 64 MB
  bf16* kh  = (bf16*)(w + 160 * MB);
  bf16* vT  = (bf16*)(w + 224 * MB);      // V^T: [2048, 16384]
  bf16* P   = (bf16*)(w + 288 * MB);      // 128 MB
  bf16* oh  = xh;

  auto cvt = [&](const float* in, bf16* out, long n) {
    long n4 = n / 4;
    cvt_f32_bf16<<<(unsigned)((n4 + 255) / 256), 256, 0, stream>>>(in, out, n4);
  };
  cvt(x,  xh,  TD);
  cvt(Wq, wqh, WW); cvt(Wk, wkh, WW); cvt(Wv, wvh, WW); cvt(Wo, woh, WW);

  const float scale = 1.0f / sqrtf((float)DMODEL);

  // Q,K projections: [16384,2048] row-major; V projection stored transposed.
  {
    dim3 g(DATTN / BN, (unsigned)(T / BM), 1);
    gemm_wmma<false, false, false><<<g, 256, 0, stream>>>(
        xh, DMODEL, wqh, DMODEL, qh, DATTN, nullptr, DATTN, DMODEL, 1.f, 0, 0, 0);
    gemm_wmma<false, false, false><<<g, 256, 0, stream>>>(
        xh, DMODEL, wkh, DMODEL, kh, DATTN, nullptr, DATTN, DMODEL, 1.f, 0, 0, 0);
    gemm_wmma<false, false, true><<<g, 256, 0, stream>>>(
        xh, DMODEL, wvh, DMODEL, vT, (int)T, nullptr, DATTN, DMODEL, 1.f, 0, 0, 0);
  }
  // logits: per batch, P[b] = scale * q[b] @ k[b]^T
  {
    dim3 g(SDIM / BN, SDIM / BM, BDIM);
    gemm_wmma<false, false, false><<<g, 256, 0, stream>>>(
        qh, DATTN, kh, DATTN, P, SDIM, nullptr, SDIM, DATTN, scale, SD, SD, SS);
  }
  softmax_rows<<<(unsigned)T, 256, 0, stream>>>(P);
  // attn_out: per batch, oh[b] = P[b] @ V[b] = P[b] @ (vT[:, b*S:(b+1)*S])^T
  {
    dim3 g(DATTN / BN, SDIM / BM, BDIM);
    gemm_wmma<false, false, false><<<g, 256, 0, stream>>>(
        P, SDIM, vT, (int)T, oh, DATTN, nullptr, DATTN, SDIM, 1.f, SS, SDIM, SD);
  }
  // output projection: out = oh @ Wo^T + bo (fp32 + bias)
  {
    dim3 g(DMODEL / BN, (unsigned)(T / BM), 1);
    gemm_wmma<true, true, false><<<g, 256, 0, stream>>>(
        oh, DATTN, woh, DMODEL, d_out, DMODEL, bo, DMODEL, DATTN, 1.f, 0, 0, 0);
  }
}
